// TransformerBlock_15358803050647
// MI455X (gfx1250) — matrix operand order repaired and checked by reading, not compiled, not run on hardware
//
#include <hip/hip_runtime.h>
#include <hip/hip_bf16.h>
#include <stdint.h>

// ---------------------------------------------------------------------------
// Types / constants
// ---------------------------------------------------------------------------
typedef __bf16 bf16_t;
typedef __attribute__((ext_vector_type(16))) __bf16 v16bf;
typedef __attribute__((ext_vector_type(8)))  __bf16 v8bf;
typedef __attribute__((ext_vector_type(8)))  float  v8f;
typedef __attribute__((ext_vector_type(4)))  unsigned int v4u;
typedef __attribute__((ext_vector_type(8)))  int v8i;
typedef __attribute__((ext_vector_type(4)))  int v4i;

#define DEV __device__ __forceinline__

static constexpr int B_  = 8;
static constexpr int C_  = 512;
static constexpr int N_  = 1024;   // H*W
static constexpr int C3_ = 1536;   // 3*C
static constexpr int C4_ = 2048;   // 4*C

#if defined(__gfx1250__) && defined(__has_builtin)
#  if __has_builtin(__builtin_amdgcn_tensor_load_to_lds) && \
      __has_builtin(__builtin_amdgcn_s_wait_tensorcnt)
#    define HAS_TDM 1
#  endif
#endif
#ifndef HAS_TDM
#  define HAS_TDM 0
#endif
// amdgpu-toolchain (clang-23 + therock headers) ships the TDM header and uses
// the 6-arg builtin form; ROCm 7.2 (clang-22) uses the 5-arg form.
#if defined(__has_include)
#  if __has_include(<hip/amd_detail/amd_gfx1250_TDM.h>)
#    define TDM_ARITY6 1
#  endif
#endif
#ifndef TDM_ARITY6
#  define TDM_ARITY6 0
#endif

DEV v8f zero8() { v8f z = {0.f,0.f,0.f,0.f,0.f,0.f,0.f,0.f}; return z; }
DEV float silu_f(float y) { return y / (1.0f + __expf(-y)); }

// ---------------------------------------------------------------------------
// WMMA fragment helpers (bf16 16x16x32, wave32). Sources are row-major
// [rows, ld] with rows contiguous along K. Works for global or LDS pointers.
// A (16x32 MxK): lane<16 -> M=lane,   K=k0+[0..7] (VGPR0-3), k0+[16..23] (VGPR4-7)
//                lane>=16-> M=lane-16,K=k0+[8..15],          k0+[24..31]
DEV v16bf load_a_frag(const bf16_t* base, int ld, int lane) {
  const int m = lane & 15, half = lane >> 4;
  const bf16_t* p = base + (size_t)m * ld + half * 8;
  v8bf lo = *(const v8bf*)(p);
  v8bf hi = *(const v8bf*)(p + 16);
  return __builtin_shufflevector(lo, hi, 0,1,2,3,4,5,6,7,8,9,10,11,12,13,14,15);
}
// B (32x16 KxN): B[k][col] = src_row[col][k]; lane<16 col=lane K=k0+[0..7],k0+[16..23],
// lane>=16 col=lane-16 K=k0+[8..15],k0+[24..31] (same order as A).
DEV v16bf load_b_frag(const bf16_t* base, int ld, int lane) {
  const int n = lane & 15, half = lane >> 4;
  const bf16_t* p = base + (size_t)n * ld + half * 8;
  v8bf lo = *(const v8bf*)(p);
  v8bf hi = *(const v8bf*)(p + 16);
  return __builtin_shufflevector(lo, hi, 0,1,2,3,4,5,6,7,8,9,10,11,12,13,14,15);
}
DEV v8f wmma_bf16(v16bf a, v16bf b, v8f c) {
  return __builtin_amdgcn_wmma_f32_16x16x32_bf16(false, a, false, b, (short)0, c,
                                                 false, false);
}

// ---------------------------------------------------------------------------
// Tensor Data Mover: stage a 2D bf16 tile [rows x rowlen] (global row stride
// `stride` elements) into contiguous LDS [rows x rowlen]. D# per ISA 8.3-8.5.
// tensor dims == tile dims so the OOB path never triggers.
// ---------------------------------------------------------------------------
DEV void tdm_load_2d(void* lds_dst, const void* gsrc, uint32_t rowlen,
                     uint32_t rows, uint32_t stride) {
#if HAS_TDM
  const uint32_t lds_off = (uint32_t)(size_t)lds_dst;   // flat addr low 32 = LDS offset
  const uint64_t ga = (uint64_t)(size_t)gsrc;
  v4u g0;
  g0[0] = 1u;                                           // count=1, user mode
  g0[1] = lds_off;                                      // lds_addr (bytes)
  g0[2] = (uint32_t)(ga & 0xffffffffu);                 // global_addr[95:64]
  g0[3] = (uint32_t)((ga >> 32) & 0x01ffffffu) | (2u << 30);  // addr hi + type=2
  uint32_t w[8] = {0u,0u,0u,0u,0u,0u,0u,0u};
  w[0] = 1u << 16;                                      // data_size=1 -> 2 bytes
  w[1] |= (rowlen & 0xffffu) << 16;                     // tensor_dim0[79:48]
  w[2] |= (rowlen >> 16) & 0xffffu;
  w[2] |= (rows & 0xffffu) << 16;                       // tensor_dim1[111:80]
  w[3] |= (rows >> 16) & 0xffffu;
  w[3] |= (rowlen & 0xffffu) << 16;                     // tile_dim0[127:112]
  w[4] |= (rows & 0xffffu);                             // tile_dim1[143:128]
  w[5] = stride;                                        // tensor_dim0_stride[207:160]
  v8i g1;
#pragma unroll
  for (int i = 0; i < 8; ++i) g1[i] = (int)w[i];
  v4i g2 = {0, 0, 0, 0}, g3 = {0, 0, 0, 0};
#if TDM_ARITY6
  v8i g4 = {0, 0, 0, 0, 0, 0, 0, 0};
  __builtin_amdgcn_tensor_load_to_lds(g0, g1, g2, g3, g4, 0);
#else
  __builtin_amdgcn_tensor_load_to_lds(g0, g1, g2, g3, 0);
#endif
#endif
  (void)lds_dst; (void)gsrc; (void)rowlen; (void)rows; (void)stride;
}
DEV void tdm_wait() {
#if HAS_TDM
  __builtin_amdgcn_s_wait_tensorcnt(0);
#endif
}
// Cooperative fallback copy (all 256 threads), 2D rows x rowlen(=mult of 8).
DEV void coop_copy_2d(bf16_t* dst, const bf16_t* src, int rows, int rowlen,
                      int stride, int tid) {
  const int cpr = rowlen >> 3;  // v8bf chunks per row
  for (int c = tid; c < rows * cpr; c += 256) {
    const int r = c / cpr, kk = (c % cpr) * 8;
    *(v8bf*)(dst + (size_t)r * rowlen + kk) =
        *(const v8bf*)(src + (size_t)r * stride + kk);
  }
}

// ---------------------------------------------------------------------------
// Shared staged mainloop: block tile = 128 A-rows x 64 B-rows, k-chunks of 64,
// double-buffered LDS fed by TDM (waves 0/1 issue DMA) or cooperative copy.
// Wave (wr,wc) computes A rows wr*32..+31  x  B rows wc*32..+31.
// ---------------------------------------------------------------------------
DEV void gemm_mainloop(const bf16_t* Arows, const bf16_t* Brows, int Cin,
                       bf16_t* sA, bf16_t* sB, int tid, v8f acc[2][2]) {
  const int lane = tid & 31, wid = tid >> 5;
  const int wr = wid & 3, wc = wid >> 2;
  const int nch = Cin >> 6;  // chunks of 64
  // prologue: stage chunk 0 into buffer 0
#if HAS_TDM
  if (wid == 0) tdm_load_2d(sA, Arows, 64, 128, (uint32_t)Cin);
  else if (wid == 1) tdm_load_2d(sB, Brows, 64, 64, (uint32_t)Cin);
  tdm_wait();
#else
  coop_copy_2d(sA, Arows, 128, 64, Cin, tid);
  coop_copy_2d(sB, Brows, 64, 64, Cin, tid);
#endif
  __syncthreads();
  for (int ch = 0; ch < nch; ++ch) {
    const int buf = ch & 1;
    if (ch + 1 < nch) {   // stage next chunk into the other buffer
      const bf16_t* an = Arows + (ch + 1) * 64;
      const bf16_t* bn = Brows + (ch + 1) * 64;
      bf16_t* da = sA + (buf ^ 1) * (128 * 64);
      bf16_t* db = sB + (buf ^ 1) * (64 * 64);
#if HAS_TDM
      if (wid == 0) tdm_load_2d(da, an, 64, 128, (uint32_t)Cin);
      else if (wid == 1) tdm_load_2d(db, bn, 64, 64, (uint32_t)Cin);
#else
      coop_copy_2d(da, an, 128, 64, Cin, tid);
      coop_copy_2d(db, bn, 64, 64, Cin, tid);
#endif
    }
    const bf16_t* ab = sA + buf * (128 * 64) + (size_t)(wr * 32) * 64;
    const bf16_t* bb = sB + buf * (64 * 64) + (size_t)(wc * 32) * 64;
#pragma unroll
    for (int kk = 0; kk < 64; kk += 32) {
      v16bf a0 = load_a_frag(ab + kk, 64, lane);
      v16bf a1 = load_a_frag(ab + 16 * 64 + kk, 64, lane);
      v16bf b0 = load_b_frag(bb + kk, 64, lane);
      v16bf b1 = load_b_frag(bb + 16 * 64 + kk, 64, lane);
      acc[0][0] = wmma_bf16(a0, b0, acc[0][0]);
      acc[0][1] = wmma_bf16(a0, b1, acc[0][1]);
      acc[1][0] = wmma_bf16(a1, b0, acc[1][0]);
      acc[1][1] = wmma_bf16(a1, b1, acc[1][1]);
    }
    if (ch + 1 < nch) tdm_wait();  // DMA of next buffer complete (issuing waves)
    __syncthreads();               // readers done + DMA visible to all waves
  }
}

// ---------------------------------------------------------------------------
// Prep kernels
// ---------------------------------------------------------------------------
__global__ __launch_bounds__(256)
void cvt_bf16_kernel(const float* __restrict__ in, bf16_t* __restrict__ out, int n) {
  int i = blockIdx.x * 256 + threadIdx.x;
  if (i < n) out[i] = (bf16_t)in[i];
}

__global__ __launch_bounds__(256)
void bn_coef_kernel(const float* __restrict__ g, const float* __restrict__ be,
                    const float* __restrict__ m, const float* __restrict__ v,
                    float* __restrict__ scale, float* __restrict__ shift, int n) {
  int i = blockIdx.x * 256 + threadIdx.x;
  if (i < n) {
    float s = g[i] * rsqrtf(v[i] + 1e-5f);
    scale[i] = s;
    shift[i] = be[i] - m[i] * s;
  }
}

// x [B,C,N] f32 -> xT [B,N,C] bf16, tiled transpose through LDS
__global__ __launch_bounds__(256)
void xpose_kernel(const float* __restrict__ x, bf16_t* __restrict__ xT) {
  __shared__ float tile[32][33];
  const int b = blockIdx.z;
  const int n0 = blockIdx.x * 32, c0 = blockIdx.y * 32;
  const float* xb = x + (size_t)b * C_ * N_;
  bf16_t* xTb = xT + (size_t)b * N_ * C_;
  const int tx = threadIdx.x, ty = threadIdx.y;
#pragma unroll
  for (int i = 0; i < 4; ++i)
    tile[ty + 8 * i][tx] = xb[(size_t)(c0 + ty + 8 * i) * N_ + n0 + tx];
  __syncthreads();
#pragma unroll
  for (int i = 0; i < 4; ++i)
    xTb[(size_t)(n0 + ty + 8 * i) * C_ + c0 + tx] = (bf16_t)tile[tx][ty + 8 * i];
}

// ---------------------------------------------------------------------------
// GEMM variant S:  D[n, o] = sum_c actT[n,c] * W[o,c]  (A rows = actT, B rows = W)
// standard-layout [O, N] output, contiguous along n per lane.
// EPI 0: += bias[o], store bf16 std                       (QKV: q,k)
// EPI 1: h = (acc + bias[o] + x)*bn1s + bn1h; f32 std + bf16 transposed
// EPI 2: out = resid + silu(acc*bn3s + bn3h); f32 std     (final)
// ---------------------------------------------------------------------------
template <int EPI>
__global__ __launch_bounds__(256)
void gemmS_kernel(const bf16_t* __restrict__ actT, const bf16_t* __restrict__ W,
                  const float* __restrict__ bias, const float* __restrict__ bnsc,
                  const float* __restrict__ bnsh, const float* __restrict__ resid,
                  float* __restrict__ out_f32, bf16_t* __restrict__ out_bf,
                  bf16_t* __restrict__ outT, int Cin, int O, int ldT) {
  __shared__ __align__(32) bf16_t sA[2 * 128 * 64];
  __shared__ __align__(32) bf16_t sB[2 * 64 * 64];
  const int tid = threadIdx.x, lane = tid & 31, wid = tid >> 5;
  const int wr = wid & 3, wc = wid >> 2;
  const int n0 = blockIdx.x * 128 + wr * 32;   // D rows (n)
  const int o0 = blockIdx.y * 64 + wc * 32;    // D cols (o)
  const int b = blockIdx.z;
  const bf16_t* actB = actT + (size_t)b * N_ * Cin;

  v8f acc[2][2] = {zero8(), zero8(), zero8(), zero8()};
  gemm_mainloop(actB + (size_t)(blockIdx.x * 128) * Cin,
                W + (size_t)(blockIdx.y * 64) * Cin, Cin, sA, sB, tid, acc);

  const int half = lane >> 4, lc = lane & 15;
  const size_t bofs = (size_t)b * O * N_;
#pragma unroll
  for (int i = 0; i < 2; ++i)
#pragma unroll
    for (int j = 0; j < 2; ++j) {
      const int o = o0 + 16 * j + lc;           // fixed per lane
      const int nb = n0 + 16 * i + 8 * half;    // 8 contiguous n
      float vals[8];
#pragma unroll
      for (int r = 0; r < 8; ++r) vals[r] = acc[i][j][r];

      if constexpr (EPI == 0) {
        const float bv = bias[o];
        v8bf ov;
#pragma unroll
        for (int r = 0; r < 8; ++r) ov[r] = (bf16_t)(vals[r] + bv);
        *(v8bf*)(out_bf + bofs + (size_t)o * N_ + nb) = ov;
      } else if constexpr (EPI == 1) {
        const float bv = bias[o], sc = bnsc[o], sh = bnsh[o];
        const v8f xr = *(const v8f*)(resid + bofs + (size_t)o * N_ + nb);
        v8f hv;
#pragma unroll
        for (int r = 0; r < 8; ++r) hv[r] = (vals[r] + bv + xr[r]) * sc + sh;
        *(v8f*)(out_f32 + bofs + (size_t)o * N_ + nb) = hv;
        bf16_t* tp = outT + (size_t)b * N_ * ldT + o;
#pragma unroll
        for (int r = 0; r < 8; ++r) tp[(size_t)(nb + r) * ldT] = (bf16_t)hv[r];
      } else {  // EPI == 2
        const float sc = bnsc[o], sh = bnsh[o];
        const v8f hr = *(const v8f*)(resid + bofs + (size_t)o * N_ + nb);
        v8f ov;
#pragma unroll
        for (int r = 0; r < 8; ++r) ov[r] = hr[r] + silu_f(vals[r] * sc + sh);
        *(v8f*)(out_f32 + bofs + (size_t)o * N_ + nb) = ov;
      }
    }
}

// ---------------------------------------------------------------------------
// GEMM variant T:  D[o, n] = sum_c W[o,c] * actT[n,c]  (A rows = W, B rows = actT)
// transposed-layout [N, ldT] output, contiguous along o per lane.
// EPI 0: += bias[o]          (QKV: v)
// EPI 1: silu(acc*bn2s+bn2h) (FFN1)
// ---------------------------------------------------------------------------
template <int EPI>
__global__ __launch_bounds__(256)
void gemmT_kernel(const bf16_t* __restrict__ W, const bf16_t* __restrict__ actT,
                  const float* __restrict__ bias, const float* __restrict__ bnsc,
                  const float* __restrict__ bnsh, bf16_t* __restrict__ outT,
                  int Cin, int O, int ldT) {
  __shared__ __align__(32) bf16_t sA[2 * 128 * 64];
  __shared__ __align__(32) bf16_t sB[2 * 64 * 64];
  const int tid = threadIdx.x, lane = tid & 31, wid = tid >> 5;
  const int wr = wid & 3, wc = wid >> 2;
  const int o0 = blockIdx.x * 128 + wr * 32;   // D rows (o)
  const int n0 = blockIdx.y * 64 + wc * 32;    // D cols (n)
  const int b = blockIdx.z;
  const bf16_t* actB = actT + (size_t)b * N_ * Cin;

  v8f acc[2][2] = {zero8(), zero8(), zero8(), zero8()};
  gemm_mainloop(W + (size_t)(blockIdx.x * 128) * Cin,
                actB + (size_t)(blockIdx.y * 64) * Cin, Cin, sA, sB, tid, acc);

  const int half = lane >> 4, lc = lane & 15;
  bf16_t* outB = outT + (size_t)b * N_ * ldT;
#pragma unroll
  for (int i = 0; i < 2; ++i)
#pragma unroll
    for (int j = 0; j < 2; ++j) {
      const int n = n0 + 16 * j + lc;           // fixed per lane
      const int ob = o0 + 16 * i + 8 * half;    // 8 contiguous o
      v8bf ov;
      if constexpr (EPI == 0) {
        const v8f bv = *(const v8f*)(bias + ob);
#pragma unroll
        for (int r = 0; r < 8; ++r) ov[r] = (bf16_t)(acc[i][j][r] + bv[r]);
      } else {  // EPI == 1: BN + SiLU
        const v8f sc = *(const v8f*)(bnsc + ob);
        const v8f sh = *(const v8f*)(bnsh + ob);
#pragma unroll
        for (int r = 0; r < 8; ++r)
          ov[r] = (bf16_t)silu_f(acc[i][j][r] * sc[r] + sh[r]);
      }
      *(v8bf*)(outB + (size_t)n * ldT + ob) = ov;
    }
}

// ---------------------------------------------------------------------------
// Attention logits + softmax rowsum.  Per (b,head): Q,K are contiguous
// [1024 x 64] blocks inside qk_std (raw-reshape identity).  The whole 128 KB
// K-head is TDM-staged into dynamic LDS once per workgroup; both passes read
// B fragments from LDS.  s_out[b*8+h, i] = sum_k softmax(Q Kt / 8)[i, k].
// ---------------------------------------------------------------------------
__global__ __launch_bounds__(256)
void attn_rowsum_kernel(const bf16_t* __restrict__ qk, float* __restrict__ s_out) {
  extern __shared__ __align__(32) bf16_t sK[];  // 1024 * 64 bf16 = 128 KB
  const int b = blockIdx.x >> 3, h = blockIdx.x & 7;
  const int tid = threadIdx.x, lane = tid & 31, wid = tid >> 5;
  const int r0 = blockIdx.y * 128 + wid * 16;
  const bf16_t* Qb = qk + ((size_t)b * (2 * C_) + h * 64) * N_;        // [1024,64]
  const bf16_t* Kb = qk + ((size_t)b * (2 * C_) + C_ + h * 64) * N_;   // [1024,64]
  const float sc = 0.125f;  // 1/sqrt(dk)

#if HAS_TDM
  if (wid == 0) tdm_load_2d(sK, Kb, 64, 1024, 64);
  tdm_wait();
#else
  coop_copy_2d(sK, Kb, 1024, 64, 64, tid);
#endif

  // A fragments for this wave's 16 rows (K = 64 -> two k-blocks), kept live.
  const v16bf a0 = load_a_frag(Qb + (size_t)r0 * 64, 64, lane);
  const v16bf a1 = load_a_frag(Qb + (size_t)r0 * 64 + 32, 64, lane);
  __syncthreads();

  float mrow[8], srow[8];
#pragma unroll
  for (int r = 0; r < 8; ++r) { mrow[r] = -3.0e38f; srow[r] = 0.f; }

  // Pass 1: online max/sum over all 64 column tiles (per-lane partials).
  for (int j = 0; j < 64; ++j) {
    v16bf b0 = load_b_frag(sK + j * 16 * 64, 64, lane);
    v16bf b1 = load_b_frag(sK + j * 16 * 64 + 32, 64, lane);
    v8f acc = zero8();
    acc = wmma_bf16(a0, b0, acc);
    acc = wmma_bf16(a1, b1, acc);
#pragma unroll
    for (int r = 0; r < 8; ++r) {
      float v = acc[r] * sc;
      float nm = fmaxf(mrow[r], v);
      srow[r] = srow[r] * __expf(mrow[r] - nm) + __expf(v - nm);
      mrow[r] = nm;
    }
  }
  // Merge per-lane partials of each row (rows r: lanes 0-15, r+8: lanes 16-31).
#pragma unroll
  for (int r = 0; r < 8; ++r) {
#pragma unroll
    for (int off = 1; off <= 8; off <<= 1) {
      float m2 = __shfl_xor(mrow[r], off, 32);
      float s2 = __shfl_xor(srow[r], off, 32);
      float nm = fmaxf(mrow[r], m2);
      srow[r] = srow[r] * __expf(mrow[r] - nm) + s2 * __expf(m2 - nm);
      mrow[r] = nm;
    }
  }
  float inv[8], tsum[8];
#pragma unroll
  for (int r = 0; r < 8; ++r) { inv[r] = 1.0f / srow[r]; tsum[r] = 0.f; }

  // Pass 2: faithful rowsum of softmax (each term divided, then summed).
  for (int j = 0; j < 64; ++j) {
    v16bf b0 = load_b_frag(sK + j * 16 * 64, 64, lane);
    v16bf b1 = load_b_frag(sK + j * 16 * 64 + 32, 64, lane);
    v8f acc = zero8();
    acc = wmma_bf16(a0, b0, acc);
    acc = wmma_bf16(a1, b1, acc);
#pragma unroll
    for (int r = 0; r < 8; ++r)
      tsum[r] += __expf(acc[r] * sc - mrow[r]) * inv[r];
  }
#pragma unroll
  for (int r = 0; r < 8; ++r) {
#pragma unroll
    for (int off = 1; off <= 8; off <<= 1) tsum[r] += __shfl_xor(tsum[r], off, 32);
  }
  if ((lane & 15) == 0) {
    const size_t base = (size_t)(b * 8 + h) * N_ + r0 + 8 * (lane >> 4);
#pragma unroll
    for (int r = 0; r < 8; ++r) s_out[base + r] = tsum[r];
  }
}

// attT[b,n,c] = vT[b,n,c] * s[b, c/64, (c%64)*16 + n/64]  (raw-reshape index map)
__global__ __launch_bounds__(256)
void vscale_kernel(const bf16_t* __restrict__ vT, const float* __restrict__ s,
                   bf16_t* __restrict__ attT) {
  const int idx = blockIdx.x * 256 + threadIdx.x;  // total 2^22
  const int b = idx >> 19;
  const int rem = idx & ((1 << 19) - 1);
  const int n = rem >> 9;
  const int c = rem & 511;
  const int head = c >> 6;
  const int i = ((c & 63) << 4) | (n >> 6);
  const float sv = s[((size_t)(b * 8 + head) << 10) + i];
  attT[idx] = (bf16_t)((float)vT[idx] * sv);
}

// ---------------------------------------------------------------------------
// Launch
// ---------------------------------------------------------------------------
extern "C" void kernel_launch(void* const* d_in, const int* in_sizes, int n_in,
                              void* d_out, int out_size, void* d_ws, size_t ws_size,
                              hipStream_t stream) {
  const float* x     = (const float*)d_in[0];
  const float* w_qkv = (const float*)d_in[1];
  const float* b_qkv = (const float*)d_in[2];
  const float* w_out = (const float*)d_in[3];
  const float* b_out = (const float*)d_in[4];
  const float* g1 = (const float*)d_in[5],  *be1 = (const float*)d_in[6];
  const float* m1 = (const float*)d_in[7],  *v1  = (const float*)d_in[8];
  const float* w_f1 = (const float*)d_in[9];
  const float* g2 = (const float*)d_in[10], *be2 = (const float*)d_in[11];
  const float* m2 = (const float*)d_in[12], *v2  = (const float*)d_in[13];
  const float* w_f2 = (const float*)d_in[14];
  const float* g3 = (const float*)d_in[15], *be3 = (const float*)d_in[16];
  const float* m3 = (const float*)d_in[17], *v3  = (const float*)d_in[18];
  float* out = (float*)d_out;

  // Workspace carve (bytes), 256B aligned.
  uint8_t* ws = (uint8_t*)d_ws;
  size_t cur = 0;
  auto carve = [&](size_t bytes) {
    void* p = ws + cur;
    cur = (cur + bytes + 255) & ~(size_t)255;
    return p;
  };
  bf16_t* wq_bf  = (bf16_t*)carve((size_t)C3_ * C_ * 2);
  bf16_t* wo_bf  = (bf16_t*)carve((size_t)C_ * C_ * 2);
  bf16_t* wf1_bf = (bf16_t*)carve((size_t)C4_ * C_ * 2);
  bf16_t* wf2_bf = (bf16_t*)carve((size_t)C_ * C4_ * 2);
  bf16_t* xT     = (bf16_t*)carve((size_t)B_ * N_ * C_ * 2);
  bf16_t* qk_std = (bf16_t*)carve((size_t)B_ * 2 * C_ * N_ * 2);
  bf16_t* vT     = (bf16_t*)carve((size_t)B_ * N_ * C_ * 2);
  bf16_t* attT   = (bf16_t*)carve((size_t)B_ * N_ * C_ * 2);
  bf16_t* hT     = (bf16_t*)carve((size_t)B_ * N_ * C_ * 2);
  bf16_t* f1T    = (bf16_t*)carve((size_t)B_ * N_ * C4_ * 2);
  float*  h_std  = (float*)carve((size_t)B_ * C_ * N_ * 4);
  float*  s_buf  = (float*)carve((size_t)B_ * 8 * N_ * 4);
  float*  bn1s = (float*)carve(C_ * 4),  *bn1h = (float*)carve(C_ * 4);
  float*  bn2s = (float*)carve(C4_ * 4), *bn2h = (float*)carve(C4_ * 4);
  float*  bn3s = (float*)carve(C_ * 4),  *bn3h = (float*)carve(C_ * 4);
  if (cur > ws_size) return;  // workspace too small; bail deterministically

  // --- Prep: weights -> bf16, BN coefficients, x -> xT bf16 ---
  cvt_bf16_kernel<<<(C3_ * C_ + 255) / 256, 256, 0, stream>>>(w_qkv, wq_bf, C3_ * C_);
  cvt_bf16_kernel<<<(C_ * C_ + 255) / 256, 256, 0, stream>>>(w_out, wo_bf, C_ * C_);
  cvt_bf16_kernel<<<(C4_ * C_ + 255) / 256, 256, 0, stream>>>(w_f1, wf1_bf, C4_ * C_);
  cvt_bf16_kernel<<<(C_ * C4_ + 255) / 256, 256, 0, stream>>>(w_f2, wf2_bf, C_ * C4_);
  bn_coef_kernel<<<(C_ + 255) / 256, 256, 0, stream>>>(g1, be1, m1, v1, bn1s, bn1h, C_);
  bn_coef_kernel<<<(C4_ + 255) / 256, 256, 0, stream>>>(g2, be2, m2, v2, bn2s, bn2h, C4_);
  bn_coef_kernel<<<(C_ + 255) / 256, 256, 0, stream>>>(g3, be3, m3, v3, bn3s, bn3h, C_);
  xpose_kernel<<<dim3(N_ / 32, C_ / 32, B_), dim3(32, 8), 0, stream>>>(x, xT);

  // --- QKV GEMM: q,k -> standard [B,2C,N] bf16 ; v -> transposed [B,N,C] bf16 ---
  gemmS_kernel<0><<<dim3(N_ / 128, (2 * C_) / 64, B_), 256, 0, stream>>>(
      xT, wq_bf, b_qkv, nullptr, nullptr, nullptr, nullptr, qk_std, nullptr,
      C_, 2 * C_, 0);
  gemmT_kernel<0><<<dim3(C_ / 128, N_ / 64, B_), 256, 0, stream>>>(
      wq_bf + (size_t)2 * C_ * C_, xT, b_qkv + 2 * C_, nullptr, nullptr, vT,
      C_, C_, C_);

  // --- Attention: logits + softmax rowsum (K-head staged in 128 KB LDS) ---
  attn_rowsum_kernel<<<dim3(B_ * 8, N_ / 128), 256, 131072, stream>>>(qk_std, s_buf);
  vscale_kernel<<<(B_ * N_ * C_) / 256, 256, 0, stream>>>(vT, s_buf, attT);

  // --- W_out GEMM + bias + residual(x) + BN1 -> h (f32 std) and hT (bf16) ---
  gemmS_kernel<1><<<dim3(N_ / 128, C_ / 64, B_), 256, 0, stream>>>(
      attT, wo_bf, b_out, bn1s, bn1h, x, h_std, nullptr, hT, C_, C_, C_);

  // --- FFN1: silu(BN2(W_f1 . h)) -> f1T [B,N,C4] bf16 ---
  gemmT_kernel<1><<<dim3(C4_ / 128, N_ / 64, B_), 256, 0, stream>>>(
      wf1_bf, hT, nullptr, bn2s, bn2h, f1T, C_, C4_, C4_);

  // --- FFN2: out = h + silu(BN3(W_f2 . f1)) -> d_out f32 [B,C,N] ---
  gemmS_kernel<2><<<dim3(N_ / 128, C_ / 64, B_), 256, 0, stream>>>(
      f1T, wf2_bf, nullptr, bn3s, bn3h, h_std, out, nullptr, nullptr,
      C4_, C_, 0);

  (void)in_sizes; (void)n_in; (void)out_size;
}
